// CausalAttention_22986664968300
// MI455X (gfx1250) — matrix operand order repaired and checked by reading, not compiled, not run on hardware
//
#include <hip/hip_runtime.h>
#include <math.h>

// ---------------- problem constants ----------------
#define B_   2
#define S_   2048
#define D_   1024
#define H_   16
#define KV_  4
#define HD_  64
#define G_   (H_ / KV_)
#define CAP_ 50.0f
#define EPS_ 1e-6f

typedef unsigned short u16;
typedef unsigned int   u32;
typedef unsigned long long u64;

typedef __attribute__((ext_vector_type(16))) __bf16 v16bf;
typedef __attribute__((ext_vector_type(8)))  float  v8f;

union AF {            // one WMMA 16-bit operand fragment: 16 bf16 per lane
    v16bf v;
    uint4 u[2];
};

// ---------------- small helpers ----------------
__device__ __forceinline__ u16 f2bf(float f) {
    union { float f; u32 u; } c; c.f = f;
    u32 r = c.u + 0x7fffu + ((c.u >> 16) & 1u);   // round-to-nearest-even
    return (u16)(r >> 16);
}
__device__ __forceinline__ float bf2f(u16 h) {
    union { u32 u; float f; } c; c.u = ((u32)h) << 16;
    return c.f;
}
__device__ __forceinline__ float half_max(float v) {   // reduce across 16-lane half
    #pragma unroll
    for (int m = 1; m <= 8; m <<= 1) v = fmaxf(v, __shfl_xor(v, m, 32));
    return v;
}
__device__ __forceinline__ float half_sum(float v) {
    #pragma unroll
    for (int m = 1; m <= 8; m <<= 1) v += __shfl_xor(v, m, 32);
    return v;
}
__device__ __forceinline__ float wave_sum(float v) {   // full wave32 reduce
    #pragma unroll
    for (int m = 1; m <= 16; m <<= 1) v += __shfl_xor(v, m, 32);
    return v;
}
// branch-free soft-cap: CAP*tanh(s/CAP) = CAP*(1 - 2/(1+exp(2s/CAP)))
__device__ __forceinline__ float softcap(float s) {
    float e = __expf(s * (2.0f / CAP_));
    return CAP_ * (1.0f - 2.0f * __builtin_amdgcn_rcpf(1.0f + e));
}
// async 16B global -> LDS copy (per active lane), tracked by ASYNCcnt
__device__ __forceinline__ void async_cp16(u32 lds_addr, const u16* g) {
    asm volatile("global_load_async_to_lds_b128 %0, %1, off"
                 :: "v"(lds_addr), "v"((u64)(__SIZE_TYPE__)g) : "memory");
}
__device__ __forceinline__ u32 lds_addr32(const void* p) {
    return (u32)(__SIZE_TYPE__)p;   // low 32 bits of generic ptr = LDS offset
}

// ---------------- fp32 -> bf16 conversion ----------------
__global__ void cvt_f32_bf16(const float* __restrict__ src, u16* __restrict__ dst, int n) {
    int i = blockIdx.x * blockDim.x + threadIdx.x;
    int stride = gridDim.x * blockDim.x;
    for (; i < n; i += stride) dst[i] = f2bf(src[i]);
}

// ---------------- generic GEMM: C(M,N) = A(M,K) @ W(N,K)^T, bf16 in, WMMA ----
// block = 128 threads = 4 waves; wave w owns rows [ (by*4+w)*16 , +16 ),
// cols [bx*64, +64) as 4 accumulators of 16x16.
// W tile (64 rows x 32 K) is shared by all 4 waves -> staged via async
// global->LDS copies, double buffered, synced with s_wait_asynccnt + barrier.
template <int OUT_BF16>
__global__ __launch_bounds__(128)
void gemm_xwt(const u16* __restrict__ A, const u16* __restrict__ W,
              void* __restrict__ Cv, int M, int N, int K) {
    const int tid  = threadIdx.x;
    const int wave = tid >> 5;
    const int lane = tid & 31;
    const int l15  = lane & 15;
    const int hi   = lane >> 4;

    const int m0 = (blockIdx.y * 4 + wave) * 16;
    const int n0 = blockIdx.x * 64;

    __shared__ __align__(16) u16 wtile[2][64 * 32];   // 2 x 4KB double buffer

    // staging plan: 64x32 bf16 tile = 256 chunks of 16B; 128 threads x 2 chunks
    const int c0   = tid;           // chunk ids tid and tid+128
    const int row0 = c0 >> 2,        col0 = (c0 & 3) * 8;
    const int c1   = tid + 128;
    const int row1 = c1 >> 2,        col1 = (c1 & 3) * 8;
    const u16* g0 = W + (size_t)(n0 + row0) * K + col0;
    const u16* g1 = W + (size_t)(n0 + row1) * K + col1;
    u32 ldsA[2], ldsB[2];
    #pragma unroll
    for (int b = 0; b < 2; ++b) {
        ldsA[b] = lds_addr32(&wtile[b][row0 * 32 + col0]);
        ldsB[b] = lds_addr32(&wtile[b][row1 * 32 + col1]);
    }

    const u16* arow = A + (size_t)(m0 + l15) * K;

    // prologue: stage first K-tile into buffer 0
    async_cp16(ldsA[0], g0);
    async_cp16(ldsB[0], g1);

    v8f acc[4] = {};
    int buf = 0;
    for (int kk = 0; kk < K; kk += 32) {
        if (kk + 32 < K) {                       // prefetch next tile (uniform)
            async_cp16(ldsA[buf ^ 1], g0 + kk + 32);
            async_cp16(ldsB[buf ^ 1], g1 + kk + 32);
            asm volatile("s_wait_asynccnt 0x2" ::: "memory");   // cur tile done
        } else {
            asm volatile("s_wait_asynccnt 0x0" ::: "memory");
        }
        __syncthreads();                         // all waves' copies visible

        AF af;                                   // A fragment: direct global
        af.u[0] = *(const uint4*)(arow + kk + 8 * hi);
        af.u[1] = *(const uint4*)(arow + kk + 16 + 8 * hi);

        const u16* wb = &wtile[buf][0];
        #pragma unroll
        for (int t = 0; t < 4; ++t) {            // B fragments from LDS
            const u16* wl = wb + (16 * t + l15) * 32 + 8 * hi;
            AF bf;
            bf.u[0] = *(const uint4*)(wl);
            bf.u[1] = *(const uint4*)(wl + 16);
            acc[t] = __builtin_amdgcn_wmma_f32_16x16x32_bf16(
                false, af.v, false, bf.v, (short)0, acc[t], false, false);
        }
        __syncthreads();                         // reads done before reuse
        buf ^= 1;
    }
    #pragma unroll
    for (int t = 0; t < 4; ++t) {
        #pragma unroll
        for (int r = 0; r < 8; ++r) {
            int   mrow = m0 + r + 8 * hi;
            int   col  = n0 + 16 * t + l15;
            size_t idx = (size_t)mrow * N + col;
            float  val = acc[t][r];
            if (OUT_BF16) ((u16*)Cv)[idx] = f2bf(val);
            else          ((float*)Cv)[idx] = val;
        }
    }
}

// ---------------- QK RMSNorm + RoPE ----------------
// one wave per (b, s, head); lane i handles pair (2i, 2i+1) of both 64-chunks.
__global__ __launch_bounds__(32)
void prep_qk(const u16* __restrict__ proj, u16* __restrict__ d1, u16* __restrict__ d2,
             const float* __restrict__ nw, const float* __restrict__ rope, int NH) {
    const int lane = threadIdx.x;
    int idx  = blockIdx.x;
    int head = idx % NH;
    int s    = (idx / NH) % S_;
    int b    = idx / (NH * S_);

    const u16* row = proj + (size_t)(b * S_ + s) * (NH * 2 * HD_) + head * (2 * HD_);

    float x0 = bf2f(row[2 * lane]),       x1 = bf2f(row[2 * lane + 1]);
    float y0 = bf2f(row[64 + 2 * lane]),  y1 = bf2f(row[64 + 2 * lane + 1]);

    float r1 = rsqrtf(wave_sum(x0 * x0 + x1 * x1) * (1.0f / HD_) + EPS_);
    float r2 = rsqrtf(wave_sum(y0 * y0 + y1 * y1) * (1.0f / HD_) + EPS_);

    float w0 = nw[2 * lane], w1 = nw[2 * lane + 1];
    x0 *= r1 * w0; x1 *= r1 * w1;
    y0 *= r2 * w0; y1 *= r2 * w1;

    float c  = rope[(s * 32 + lane) * 2 + 0];
    float sn = rope[(s * 32 + lane) * 2 + 1];
    float o0 = x0 * c - x1 * sn, o1 = x0 * sn + x1 * c;
    float p0 = y0 * c - y1 * sn, p1 = y0 * sn + y1 * c;

    size_t o = ((size_t)(b * NH + head) * S_ + s) * HD_ + 2 * lane;
    d1[o] = f2bf(o0); d1[o + 1] = f2bf(o1);
    d2[o] = f2bf(p0); d2[o + 1] = f2bf(p1);
}

// ---------------- V transpose: [b*s][kv*64+d] -> [b][kv][d][s] -------------
__global__ void vtrans(const u16* __restrict__ vp, u16* __restrict__ vt, int total) {
    int i = blockIdx.x * blockDim.x + threadIdx.x;
    if (i >= total) return;
    int s   = i % S_;
    int d   = (i / S_) % HD_;
    int bkv = i / (S_ * HD_);
    int b   = bkv / KV_;
    int kv  = bkv % KV_;
    vt[i] = vp[(size_t)(b * S_ + s) * (KV_ * HD_) + kv * HD_ + d];
}

// ---------------- flash differential attention ----------------
__device__ __forceinline__ void softmax_step(
    v8f s0acc, v8f s1acc, float* m, float* l, v8f* o, u16* pld,
    int q0, int kb, int l15, int hi) {
    const float scale = 0.125f;          // 1/sqrt(64)
    #pragma unroll
    for (int r = 0; r < 8; ++r) {
        float a = softcap(s0acc[r] * scale);
        float b = softcap(s1acc[r] * scale);
        int qm = q0 + r + 8 * hi;
        if (kb + l15 > qm)      a = -3.0e38f;
        if (kb + 16 + l15 > qm) b = -3.0e38f;
        float mn = fmaxf(m[r], half_max(fmaxf(a, b)));
        float al = __expf(m[r] - mn);
        float p0 = __expf(a - mn);
        float p1 = __expf(b - mn);
        l[r] = l[r] * al + half_sum(p0 + p1);
        m[r] = mn;
        pld[(r + 8 * hi) * 32 + l15]      = f2bf(p0);
        pld[(r + 8 * hi) * 32 + 16 + l15] = f2bf(p1);
        #pragma unroll
        for (int t = 0; t < 4; ++t) o[t][r] *= al;
    }
}

__global__ __launch_bounds__(32)
void attn_kernel(const u16* __restrict__ q1b, const u16* __restrict__ q2b,
                 const u16* __restrict__ k1b, const u16* __restrict__ k2b,
                 const u16* __restrict__ vtb, const float* __restrict__ lam_p,
                 u16* __restrict__ ao) {
    const int lane = threadIdx.x;
    const int l15  = lane & 15;
    const int hi   = lane >> 4;
    const int q0   = blockIdx.x * 16;
    const int bh   = blockIdx.y;
    const int b    = bh / H_;
    const int h    = bh % H_;
    const int kv   = h / G_;

    __shared__ __align__(16) u16 pld1[16 * 32];
    __shared__ __align__(16) u16 pld2[16 * 32];

    // Q fragments (row-major A-operand: two 16B chunks per lane per K-step)
    const u16* q1row = q1b + ((size_t)(b * H_ + h) * S_ + q0 + l15) * HD_;
    const u16* q2row = q2b + ((size_t)(b * H_ + h) * S_ + q0 + l15) * HD_;
    AF q1f[2], q2f[2];
    #pragma unroll
    for (int f = 0; f < 2; ++f) {
        int kk = 32 * f;
        q1f[f].u[0] = *(const uint4*)(q1row + kk + 8 * hi);
        q1f[f].u[1] = *(const uint4*)(q1row + kk + 16 + 8 * hi);
        q2f[f].u[0] = *(const uint4*)(q2row + kk + 8 * hi);
        q2f[f].u[1] = *(const uint4*)(q2row + kk + 16 + 8 * hi);
    }

    v8f o1[4] = {}, o2[4] = {};
    float m1[8], l1[8], m2[8], l2[8];
    #pragma unroll
    for (int r = 0; r < 8; ++r) { m1[r] = -INFINITY; m2[r] = -INFINITY; l1[r] = 0.f; l2[r] = 0.f; }

    const size_t kbase = (size_t)(b * KV_ + kv) * S_;
    const size_t vbase = (size_t)(b * KV_ + kv) * HD_;

    for (int kb = 0; kb <= q0 + 15; kb += 32) {
        // ---- scores: S1 = Q1 K1^T, S2 = Q2 K2^T (16x32 each) ----
        v8f s1[2] = {}, s2[2] = {};
        #pragma unroll
        for (int t = 0; t < 2; ++t) {
            const u16* k1row = k1b + (kbase + kb + 16 * t + l15) * HD_;
            const u16* k2row = k2b + (kbase + kb + 16 * t + l15) * HD_;
            #pragma unroll
            for (int f = 0; f < 2; ++f) {
                int kk = 32 * f;
                AF b1, b2;
                b1.u[0] = *(const uint4*)(k1row + kk + 8 * hi);
                b1.u[1] = *(const uint4*)(k1row + kk + 16 + 8 * hi);
                b2.u[0] = *(const uint4*)(k2row + kk + 8 * hi);
                b2.u[1] = *(const uint4*)(k2row + kk + 16 + 8 * hi);
                s1[t] = __builtin_amdgcn_wmma_f32_16x16x32_bf16(
                    false, q1f[f].v, false, b1.v, (short)0, s1[t], false, false);
                s2[t] = __builtin_amdgcn_wmma_f32_16x16x32_bf16(
                    false, q2f[f].v, false, b2.v, (short)0, s2[t], false, false);
            }
        }

        // ---- dual online softmax + stage P tiles in LDS ----
        softmax_step(s1[0], s1[1], m1, l1, o1, pld1, q0, kb, l15, hi);
        softmax_step(s2[0], s2[1], m2, l2, o2, pld2, q0, kb, l15, hi);

        asm volatile("s_wait_dscnt 0x0" ::: "memory");

        AF p1f, p2f;
        p1f.u[0] = *(const uint4*)(pld1 + l15 * 32 + 8 * hi);
        p1f.u[1] = *(const uint4*)(pld1 + l15 * 32 + 16 + 8 * hi);
        p2f.u[0] = *(const uint4*)(pld2 + l15 * 32 + 8 * hi);
        p2f.u[1] = *(const uint4*)(pld2 + l15 * 32 + 16 + 8 * hi);

        // ---- O += P @ V (V^T layout -> contiguous B-fragment loads) ----
        #pragma unroll
        for (int t = 0; t < 4; ++t) {
            const u16* vrow = vtb + (vbase + 16 * t + l15) * S_ + kb + 8 * hi;
            AF vf;
            vf.u[0] = *(const uint4*)(vrow);
            vf.u[1] = *(const uint4*)(vrow + 16);
            o1[t] = __builtin_amdgcn_wmma_f32_16x16x32_bf16(
                false, p1f.v, false, vf.v, (short)0, o1[t], false, false);
            o2[t] = __builtin_amdgcn_wmma_f32_16x16x32_bf16(
                false, p2f.v, false, vf.v, (short)0, o2[t], false, false);
        }
    }

    const float lam = *lam_p;
    #pragma unroll
    for (int t = 0; t < 4; ++t) {
        #pragma unroll
        for (int r = 0; r < 8; ++r) {
            float inv1 = __builtin_amdgcn_rcpf(l1[r]);
            float inv2 = __builtin_amdgcn_rcpf(l2[r]);
            float val  = o1[t][r] * inv1 - lam * (o2[t][r] * inv2);
            int   row = q0 + r + 8 * hi;
            int   col = h * HD_ + 16 * t + l15;
            ao[(size_t)(b * S_ + row) * (H_ * HD_) + col] = f2bf(val);
        }
    }
}

// ---------------- launch ----------------
extern "C" void kernel_launch(void* const* d_in, const int* in_sizes, int n_in,
                              void* d_out, int out_size, void* d_ws, size_t ws_size,
                              hipStream_t stream) {
    const float* x    = (const float*)d_in[0];
    const float* rope = (const float*)d_in[1];
    const float* wq   = (const float*)d_in[2];
    const float* wk   = (const float*)d_in[3];
    const float* wv   = (const float*)d_in[4];
    const float* wo   = (const float*)d_in[5];
    const float* qnw  = (const float*)d_in[6];
    const float* knw  = (const float*)d_in[7];
    const float* lam  = (const float*)d_in[8];
    float* out = (float*)d_out;

    const int M = B_ * S_;                 // 4096 rows
    // workspace carve-up (bf16 elements -> bytes), 256B aligned
    char* ws = (char*)d_ws;
    size_t off = 0;
    auto take = [&](size_t elems) {
        u16* p = (u16*)(ws + off);
        off = (off + elems * sizeof(u16) + 255) & ~(size_t)255;
        return p;
    };
    u16* xb  = take((size_t)M * D_);
    u16* wqb = take((size_t)2 * H_ * HD_ * D_);
    u16* wkb = take((size_t)2 * KV_ * HD_ * D_);
    u16* wvb = take((size_t)KV_ * HD_ * D_);
    u16* wob = take((size_t)D_ * H_ * HD_);
    u16* qp  = take((size_t)M * 2 * H_ * HD_);
    u16* kp  = take((size_t)M * 2 * KV_ * HD_);
    u16* vp  = take((size_t)M * KV_ * HD_);
    u16* q1b = take((size_t)B_ * H_ * S_ * HD_);
    u16* q2b = take((size_t)B_ * H_ * S_ * HD_);
    u16* k1b = take((size_t)B_ * KV_ * S_ * HD_);
    u16* k2b = take((size_t)B_ * KV_ * S_ * HD_);
    u16* vtb = take((size_t)B_ * KV_ * HD_ * S_);
    u16* ao  = take((size_t)M * H_ * HD_);

    auto cvt = [&](const float* s, u16* d, int n) {
        int blocks = (n + 255) / 256;
        if (blocks > 4096) blocks = 4096;
        cvt_f32_bf16<<<blocks, 256, 0, stream>>>(s, d, n);
    };
    cvt(x,  xb,  M * D_);
    cvt(wq, wqb, 2 * H_ * HD_ * D_);
    cvt(wk, wkb, 2 * KV_ * HD_ * D_);
    cvt(wv, wvb, KV_ * HD_ * D_);
    cvt(wo, wob, D_ * H_ * HD_);

    // QKV projections (bf16 out)
    gemm_xwt<1><<<dim3((2 * H_ * HD_) / 64, M / 64), 128, 0, stream>>>(
        xb, wqb, qp, M, 2 * H_ * HD_, D_);
    gemm_xwt<1><<<dim3((2 * KV_ * HD_) / 64, M / 64), 128, 0, stream>>>(
        xb, wkb, kp, M, 2 * KV_ * HD_, D_);
    gemm_xwt<1><<<dim3((KV_ * HD_) / 64, M / 64), 128, 0, stream>>>(
        xb, wvb, vp, M, KV_ * HD_, D_);

    // RMSNorm + RoPE for q1/q2 and k1/k2
    prep_qk<<<B_ * S_ * H_,  32, 0, stream>>>(qp, q1b, q2b, qnw, rope, H_);
    prep_qk<<<B_ * S_ * KV_, 32, 0, stream>>>(kp, k1b, k2b, knw, rope, KV_);

    // V transpose
    {
        int total = B_ * KV_ * HD_ * S_;
        vtrans<<<(total + 255) / 256, 256, 0, stream>>>(vp, vtb, total);
    }

    // flash differential attention
    attn_kernel<<<dim3(S_ / 16, B_ * H_), 32, 0, stream>>>(
        q1b, q2b, k1b, k2b, vtb, lam, ao);

    // output projection (fp32 out)
    gemm_xwt<0><<<dim3(D_ / 64, M / 64), 128, 0, stream>>>(
        ao, wob, out, M, D_, H_ * HD_);
}
